// EMA_25039659336445
// MI455X (gfx1250) — matrix operand order repaired and checked by reading, not compiled, not run on hardware
//
#include <hip/hip_runtime.h>

typedef __attribute__((ext_vector_type(16))) _Float16 v16h;
typedef __attribute__((ext_vector_type(4)))  _Float16 v4h;
typedef __attribute__((ext_vector_type(8)))  float    v8f;

#define THREADS 512
#define IC      32
#define HW      4096
#define GXS     36          /* halves per pixel slot: 32 ch + 4 pad (72B, 8B-aligned) */
#define NBG_CH  (IC * HW)
#define NSLOTS  (64 * 65 + 1)   /* 64 rows x 65 cols (col 64 = zero halo) + 1 front zero slot */
#define PSLOT(y, x) (1 + (y) * 65 + (x))

// LDS region byte offsets (dynamic shared memory)
#define SM_GX_BYTES    (NSLOTS * GXS * 2)          /* 299592 */
#define SM_ATTN_OFF    299600                      /* 16-aligned; 4096 f32 attn1 / temps */
#define SM_MASK_OFF    (SM_ATTN_OFF + 16384)       /* 2 x 64 x 32 f16 masks */
#define SM_SCR_OFF     (SM_MASK_OFF + 8192)        /* small scratch */
#define SM_TOTAL       (SM_SCR_OFF + 1024)         /* = 325200 < 327680 (320 KB) */

__device__ __forceinline__ float sigm(float v) { return 1.0f / (1.0f + __expf(-v)); }

// Branchless B-operand fetch: 16 consecutive halves (one K-half of a 32x16 B tile column)
__device__ __forceinline__ v16h ldB(const _Float16* p) {
    const v4h* s = (const v4h*)p;
    v4h a = s[0], b = s[1], c = s[2], d = s[3];
    v16h r;
    #pragma unroll
    for (int j = 0; j < 4; ++j) {
        r[j] = a[j]; r[4 + j] = b[j]; r[8 + j] = c[j]; r[12 + j] = d[j];
    }
    return r;
}

extern "C" __global__ __launch_bounds__(THREADS)
void cc_gate_kernel(const float* __restrict__ x,
                    const float* __restrict__ sf_w, const float* __restrict__ sf_b,
                    const float* __restrict__ lc_w, const float* __restrict__ lc_b,
                    const float* __restrict__ gn_w, const float* __restrict__ gn_b,
                    float* __restrict__ out)
{
    extern __shared__ __align__(16) char smem[];
    _Float16* gxh   = (_Float16*)smem;
    float*    attn1 = (float*)(smem + SM_ATTN_OFF);
    _Float16* hm    = (_Float16*)(smem + SM_MASK_OFF);        // [row][ch]
    _Float16* wm    = hm + 64 * IC;                           // [col][ch]
    float*    persum = (float*)(smem + SM_SCR_OFF);           // [32]
    float*    scaleB = persum + 32;
    float*    biasB  = scaleB + 32;
    float*    awg    = biasB + 32;
    float*    carr   = awg + 32;
    float*    consts = carr + 32;

    const int tid  = threadIdx.x;
    const int bg   = blockIdx.x;
    const size_t base = (size_t)bg * NBG_CH;

    // ---- Phase A: zero halo slots, then stream image global->LDS f16 (float4 reads) ----
    for (int t = tid; t < 65; t += THREADS) {
        int slot = (t == 64) ? 0 : (PSLOT(t, 64));
        #pragma unroll
        for (int c = 0; c < IC; ++c) gxh[slot * GXS + c] = (_Float16)0.f;
    }
    {
        const float4* x4 = (const float4*)(x + base);
        for (int i = tid; i < NBG_CH / 4; i += THREADS) {
            float4 v = x4[i];
            int idx = i * 4;
            int ch = idx >> 12;
            int p  = idx & (HW - 1);
            int y = p >> 6, xx = p & 63;              // xx <= 60: all 4 stay in-row
            int s = PSLOT(y, xx) * GXS + ch;
            gxh[s]           = (_Float16)v.x;
            gxh[s + GXS]     = (_Float16)v.y;
            gxh[s + 2 * GXS] = (_Float16)v.z;
            gxh[s + 3 * GXS] = (_Float16)v.w;
        }
    }
    __syncthreads();

    // ---- Phase B: hp (mean over x) / wp (mean over y) into attn1 temp ----
    for (int t = tid; t < 2 * IC * 64; t += THREADS) {
        int which = t >> 11;
        int ch = (t >> 6) & 31;
        int rc = t & 63;
        float s = 0.f;
        if (which == 0) {
            for (int xx = 0; xx < 64; ++xx) s += (float)gxh[PSLOT(rc, xx) * GXS + ch];
        } else {
            for (int yy = 0; yy < 64; ++yy) s += (float)gxh[PSLOT(yy, rc) * GXS + ch];
        }
        attn1[t] = s * (1.0f / 64.0f);
    }
    __syncthreads();

    // ---- Phase C: 1x1 conv over concat(hp,wp), sigmoid -> masks (f16, transposed) ----
    for (int t = tid; t < IC * 128; t += THREADS) {
        int o = t >> 7, l = t & 127;
        float acc = sf_b[o];
        for (int i = 0; i < IC; ++i) {
            float c = (l < 64) ? attn1[i * 64 + l] : attn1[2048 + i * 64 + (l - 64)];
            acc += sf_w[o * IC + i] * c;
        }
        float mval = sigm(acc);
        if (l < 64) hm[l * IC + o] = (_Float16)mval;
        else        wm[(l - 64) * IC + o] = (_Float16)mval;
    }
    __syncthreads();

    // ---- Phase D: instance-norm stats of mod = gx*hm*wm (16 threads/ch) ----
    {
        int sub = tid & 15, ch = tid >> 4;
        float s1 = 0.f, s2 = 0.f;
        for (int q = 0; q < 256; ++q) {
            int p = sub * 256 + q;
            int y = p >> 6, xx = p & 63;
            float g = (float)gxh[PSLOT(y, xx) * GXS + ch];
            float m = g * (float)hm[y * IC + ch] * (float)wm[xx * IC + ch];
            s1 += m; s2 += m * m;
        }
        attn1[ch * 16 + sub]       = s1;
        attn1[512 + ch * 16 + sub] = s2;
    }
    __syncthreads();
    if (tid < IC) {
        float s1 = 0.f, s2 = 0.f;
        for (int k = 0; k < 16; ++k) { s1 += attn1[tid * 16 + k]; s2 += attn1[512 + tid * 16 + k]; }
        float mean = s1 * (1.0f / HW);
        float var  = s2 * (1.0f / HW) - mean * mean;
        float sc   = gn_w[tid] * rsqrtf(var + 1e-5f);
        scaleB[tid] = sc;
        biasB[tid]  = gn_b[tid] - mean * sc;
    }
    if (tid == 0) {  // aw_gn = softmax(gn_b): spatial mean of gn == gn_b exactly
        float mx = -1e30f;
        for (int i = 0; i < IC; ++i) mx = fmaxf(mx, gn_b[i]);
        float s = 0.f;
        for (int i = 0; i < IC; ++i) { float e = __expf(gn_b[i] - mx); awg[i] = e; s += e; }
        float inv = 1.0f / s;
        for (int i = 0; i < IC; ++i) awg[i] *= inv;
    }
    __syncthreads();
    for (int p = tid; p < HW; p += THREADS) attn1[p] = 0.f;
    if (tid < IC) persum[tid] = 0.f;
    __syncthreads();

    // ---- Phase E: 3x3 conv via WMMA f32_16x16x32_f16, streaming epilogue ----
    const int lane  = tid & 31;
    const int wv    = tid >> 5;         // 16 waves
    const int m     = lane & 15;        // A: o-row / B: pixel column
    const int khalf = lane >> 4;        // K-half select (contiguous K order)

    for (int pass = 0; pass < 2; ++pass) {
        const int obase = pass * 16;
        const int o = obase + m;
        v16h A[9];
        #pragma unroll
        for (int tap = 0; tap < 9; ++tap) {
            int ky = tap / 3, kx = tap % 3;
            v16h a;
            #pragma unroll
            for (int j = 0; j < 16; ++j) {
                int pos = j & 7, half = j >> 3;
                int k = j + khalf * 16;   // input channel (contiguous K order, matches B)
                a[j] = (_Float16)lc_w[((o * IC + k) * 3 + ky) * 3 + kx];
            }
            A[tap] = a;
        }
        float aw8[8];
        #pragma unroll
        for (int r = 0; r < 8; ++r) aw8[r] = awg[obase + r + khalf * 8];

        v8f psum = {};
        for (int ti = 0; ti < 16; ++ti) {           // 16 pixel-tiles per wave
            int t = wv * 16 + ti;
            int y = t >> 2, x0 = (t & 3) * 16;
            v8f C = {};
            #pragma unroll
            for (int ky = 0; ky < 3; ++ky) {
                int yy = y + ky - 1;
                if (yy >= 0 && yy < 64) {           // wave-uniform; zero rows skipped exactly
                    // branchless via halo: px in [-1,64] all map to valid (zeroed) slots
                    const _Float16* rp = gxh + (1 + yy * 65 + (x0 + m - 1)) * GXS + khalf * 16;
                    v16h B0 = ldB(rp);
                    v16h B1 = ldB(rp + GXS);
                    v16h B2 = ldB(rp + 2 * GXS);
                    C = __builtin_amdgcn_wmma_f32_16x16x32_f16(
                            false, A[ky * 3 + 0], false, B0, (short)0, C, false, false);
                    C = __builtin_amdgcn_wmma_f32_16x16x32_f16(
                            false, A[ky * 3 + 1], false, B1, (short)0, C, false, false);
                    C = __builtin_amdgcn_wmma_f32_16x16x32_f16(
                            false, A[ky * 3 + 2], false, B2, (short)0, C, false, false);
                }
            }
            psum += C;
            float part = 0.f;
            #pragma unroll
            for (int r = 0; r < 8; ++r) part += aw8[r] * C[r];
            part += __shfl_xor(part, 16, 32);        // combine the two o-row halves
            if (lane < 16) attn1[y * 64 + x0 + lane] += part;   // wave-exclusive pixels
        }
        #pragma unroll
        for (int r = 0; r < 8; ++r)
            atomicAdd(&persum[obase + r + khalf * 8], psum[r]);
    }
    __syncthreads();

    // ---- Phase F: aw_local = softmax(mean(local)); fold constants ----
    if (tid == 0) {
        float ml[IC], awl[IC];
        float mx = -1e30f;
        for (int i = 0; i < IC; ++i) { ml[i] = persum[i] * (1.0f / HW) + lc_b[i]; mx = fmaxf(mx, ml[i]); }
        float s = 0.f;
        for (int i = 0; i < IC; ++i) { awl[i] = __expf(ml[i] - mx); s += awl[i]; }
        float inv = 1.0f / s, c1 = 0.f, c2 = 0.f;
        for (int i = 0; i < IC; ++i) {
            awl[i] *= inv;
            carr[i] = awl[i] * scaleB[i];
            c2 += awl[i] * biasB[i];
            c1 += awg[i] * lc_b[i];          // local-bias part of attn1
        }
        consts[0] = c1 + c2;
    }
    __syncthreads();

    // ---- Phase G: attn2 on the fly, gate, scale fresh fp32 input ----
    for (int p = tid; p < HW; p += THREADS) {
        int y = p >> 6, xx = p & 63;
        int slot = PSLOT(y, xx) * GXS;
        float s = attn1[p] + consts[0];
        for (int i = 0; i < IC; ++i)
            s += carr[i] * (float)gxh[slot + i]
                         * (float)hm[y * IC + i] * (float)wm[xx * IC + i];
        float gate = sigm(s);
        for (int ch = 0; ch < IC; ++ch) {
            size_t idx = base + (size_t)ch * HW + p;
            out[idx] = x[idx] * gate;      // coalesced across threads per ch
        }
    }
}

extern "C" void kernel_launch(void* const* d_in, const int* in_sizes, int n_in,
                              void* d_out, int out_size, void* d_ws, size_t ws_size,
                              hipStream_t stream) {
    (void)n_in; (void)out_size; (void)d_ws; (void)ws_size;
    const float* x    = (const float*)d_in[0];
    const float* sf_w = (const float*)d_in[1];
    const float* sf_b = (const float*)d_in[2];
    const float* lc_w = (const float*)d_in[3];
    const float* lc_b = (const float*)d_in[4];
    const float* gn_w = (const float*)d_in[5];
    const float* gn_b = (const float*)d_in[6];
    float* out = (float*)d_out;

    int nbg = in_sizes[0] / NBG_CH;   // 256 for the reference shapes
    cc_gate_kernel<<<dim3(nbg), dim3(THREADS), SM_TOTAL, stream>>>(
        x, sf_w, sf_b, lc_w, lc_b, gn_w, gn_b, out);
}
